// MultiHeadAttention_39573828665503
// MI455X (gfx1250) — matrix operand order repaired and checked by reading, not compiled, not run on hardware
//
#include <hip/hip_runtime.h>
#include <math.h>

// ---------------------------------------------------------------------------
// Problem sizes (fixed by the reference)
// ---------------------------------------------------------------------------
#define BB   4
#define LL   1024
#define DD   1024
#define HH   16
#define HD   64          // head dim
#define NEGV (-1e9f)

typedef __attribute__((ext_vector_type(16))) __bf16 v16bf;
typedef __attribute__((ext_vector_type(8)))  __bf16 v8bf;
typedef __attribute__((ext_vector_type(8)))  float  v8f;

// fp32 -> bf16 round-to-nearest-even (pure bit math)
static __device__ __forceinline__ unsigned short bfbits(float f) {
  unsigned u = __builtin_bit_cast(unsigned, f);
  u += 0x7fffu + ((u >> 16) & 1u);
  return (unsigned short)(u >> 16);
}
static __device__ __forceinline__ __bf16 f2bf(float f) {
  unsigned short h = bfbits(f);
  return __builtin_bit_cast(__bf16, h);
}

// ---------------------------------------------------------------------------
// WMMA fragment loaders (wave32, same contiguous k order for A and B), bf16 sources
//
// A (16x32): lane m = lane&15, half = lane>>4.
//   e=0..15 -> K = k0 + half*16 + e (contiguous, same as B)
// B (32x16): lane n = lane&15, half = lane>>4 ; e -> K = k0 + half*16 + e
// C/D (16x16 f32): lane n = lane&15; vgpr r -> M = r + 8*(lane>>4)
// ---------------------------------------------------------------------------
static __device__ __forceinline__ v16bf load_a_bf16(const __bf16* __restrict__ base,
                                                    int ld, int m0, int k0, int lane) {
  const int m = lane & 15, half = lane >> 4;
  const __bf16* p = base + (size_t)(m0 + m) * ld + k0 + half * 16;
  v8bf lo = *(const v8bf*)(p);        // 16B aligned
  v8bf hi = *(const v8bf*)(p + 8);
  return __builtin_shufflevector(lo, hi, 0, 1, 2, 3, 4, 5, 6, 7,
                                 8, 9, 10, 11, 12, 13, 14, 15);
}

// B = X^T with X row-major bf16 (N rows x K cols): 16 contiguous K per lane
static __device__ __forceinline__ v16bf load_bt_bf16(const __bf16* __restrict__ base,
                                                     int ld, int n0, int k0, int lane) {
  const int n = lane & 15, half = lane >> 4;
  const __bf16* p = base + (size_t)(n0 + n) * ld + k0 + half * 16;
  return *(const v16bf*)(p);          // 32B aligned
}

static __device__ __forceinline__ void store_tile_f32(float* __restrict__ base, int ld,
                                                      int m0, int n0, int lane, v8f c) {
  const int n = lane & 15, half = lane >> 4;
  float* p = base + (size_t)(m0 + half * 8) * ld + n0 + n;
#pragma unroll
  for (int r = 0; r < 8; ++r) p[(size_t)r * ld] = c[r];
}

static __device__ __forceinline__ void store_tile_bf16(__bf16* __restrict__ base, int ld,
                                                       int m0, int n0, int lane, v8f c) {
  const int n = lane & 15, half = lane >> 4;
  __bf16* p = base + (size_t)(m0 + half * 8) * ld + n0 + n;
#pragma unroll
  for (int r = 0; r < 8; ++r) p[(size_t)r * ld] = f2bf(c[r]);
}

#define WMMA_BF16(a, b, c) \
  __builtin_amdgcn_wmma_f32_16x16x32_bf16(false, (a), false, (b), (short)0, (c), false, false)

// ---------------------------------------------------------------------------
// Elementwise fp32 -> bf16 (row-major copy); one float4 per thread.
// ---------------------------------------------------------------------------
__global__ __launch_bounds__(256) void cvt_f32_bf16(const float* __restrict__ in,
                                                    __bf16* __restrict__ out) {
  const size_t i = (size_t)blockIdx.x * 256 + threadIdx.x;
  float4 f = ((const float4*)in)[i];
  uint2 o;
  o.x = (unsigned)bfbits(f.x) | ((unsigned)bfbits(f.y) << 16);
  o.y = (unsigned)bfbits(f.z) | ((unsigned)bfbits(f.w) << 16);
  ((uint2*)out)[i] = o;
}

// ---------------------------------------------------------------------------
// Pack a row-major fp32 K x N matrix into per-tile WMMA B-fragment layout:
// tile (kt, nt) -> out[tile*512 + lane*16 + e] = W[(kt*32 + half*16 + e)*N + nt*16 + n]
// One wave packs one 32x16 tile; 8 waves per block.
// ---------------------------------------------------------------------------
__global__ __launch_bounds__(256) void pack_b(const float* __restrict__ W,
                                              __bf16* __restrict__ out, int N) {
  const int lane = threadIdx.x & 31, wave = threadIdx.x >> 5;
  const int tile = blockIdx.x * 8 + wave;
  const int ntn = N >> 4;
  const int kt = tile / ntn, nt = tile - kt * ntn;
  const int n = lane & 15, half = lane >> 4;
  const float* p = W + (size_t)(kt * 32 + half * 16) * N + nt * 16 + n;
  v16bf v;
#pragma unroll
  for (int e = 0; e < 16; ++e) v[e] = f2bf(p[(size_t)e * N]);
  *(v16bf*)(out + (size_t)tile * 512 + lane * 16) = v;
}

// ---------------------------------------------------------------------------
// GEMM: C[M x 1024] = A[M x 1024] * W[1024 x 1024]
// A: bf16 row-major. Bp: packed B tiles. C: fp32 (+ optional bf16 copy).
// 256 threads = 8 waves arranged 2x4; each wave owns a 32x32 register tile
// (2x2 WMMA tiles) -> block tile 64x128. Per k-step: 2 A-frags + 2 B-frags
// feed 4 WMMAs (2 loads per WMMA instead of 4).
// ---------------------------------------------------------------------------
__global__ __launch_bounds__(256) void gemm_bf16p(const __bf16* __restrict__ A,
                                                  const __bf16* __restrict__ Bp,
                                                  float* __restrict__ C,
                                                  __bf16* __restrict__ Cbf) {
  const int lane = threadIdx.x & 31, wave = threadIdx.x >> 5;
  const int m0 = blockIdx.y * 64 + (wave >> 2) * 32;
  const int n0 = blockIdx.x * 128 + (wave & 3) * 32;
  const int nt = n0 >> 4;
  v8f acc00 = {}, acc01 = {}, acc10 = {}, acc11 = {};
  for (int kt = 0; kt < 32; ++kt) {
    const int k0 = kt * 32;
    v16bf a0 = load_a_bf16(A, DD, m0, k0, lane);
    v16bf a1 = load_a_bf16(A, DD, m0 + 16, k0, lane);
    v16bf b0 = *(const v16bf*)(Bp + ((size_t)(kt * 64 + nt) * 512 + lane * 16));
    v16bf b1 = *(const v16bf*)(Bp + ((size_t)(kt * 64 + nt + 1) * 512 + lane * 16));
    __builtin_prefetch(Bp + ((size_t)((kt + 1) * 64 + nt) * 512 + lane * 16), 0, 1);
    acc00 = WMMA_BF16(a0, b0, acc00);
    acc01 = WMMA_BF16(a0, b1, acc01);
    acc10 = WMMA_BF16(a1, b0, acc10);
    acc11 = WMMA_BF16(a1, b1, acc11);
  }
  store_tile_f32(C, DD, m0, n0, lane, acc00);
  store_tile_f32(C, DD, m0, n0 + 16, lane, acc01);
  store_tile_f32(C, DD, m0 + 16, n0, lane, acc10);
  store_tile_f32(C, DD, m0 + 16, n0 + 16, lane, acc11);
  if (Cbf) {
    store_tile_bf16(Cbf, DD, m0, n0, lane, acc00);
    store_tile_bf16(Cbf, DD, m0, n0 + 16, lane, acc01);
    store_tile_bf16(Cbf, DD, m0 + 16, n0, lane, acc10);
    store_tile_bf16(Cbf, DD, m0 + 16, n0 + 16, lane, acc11);
  }
}

// ---------------------------------------------------------------------------
// Fused attention: scores (WMMA) + dual mask + softmax + attn store (fp32) +
// ctx = probs @ V_head (WMMA, probs staged as bf16 in LDS) -> ctx bf16.
// One block = one (b,h) x 16-row q strip. 8 waves.
//   Phase 1: wave w owns score tiles at columns (w + 8t)*16, t=0..7 (regs).
//   Phase 4: wave w owns ctx col tile (w&3), K-half (w>>2)*512; partials
//            combined via LDS.
// ---------------------------------------------------------------------------
__global__ __launch_bounds__(256) void attn_fused(const __bf16* __restrict__ qpbf,
                                                  const __bf16* __restrict__ kpbf,
                                                  const int* __restrict__ mask1,
                                                  const int* __restrict__ mask2,
                                                  float* __restrict__ attn,
                                                  const __bf16* __restrict__ vpk,
                                                  __bf16* __restrict__ ctxbf) {
  const int lane = threadIdx.x & 31, wave = threadIdx.x >> 5;
  const int half = lane >> 4, nlane = lane & 15;
  const int bh = blockIdx.y, b = bh / HH, h = bh % HH;
  const int q0 = blockIdx.x * 16;
  const __bf16* qb = qpbf + (size_t)b * LL * DD + h * HD;
  const __bf16* kb = kpbf + (size_t)b * LL * DD + h * HD;
  float* ab = attn + (size_t)bh * LL * LL;

  __shared__ __bf16 sprob[16 * LL];       // 32KB: prob strip, bf16 row-major
  __shared__ float wmax[8][16];
  __shared__ float wsum[8][16];
  __shared__ float cred[4][16][16];       // 4KB: ctx partial reduction

  // ---- Phase 1: scores -------------------------------------------------
  v16bf a0 = load_a_bf16(qb, DD, q0, 0, lane);
  v16bf a1 = load_a_bf16(qb, DD, q0, 32, lane);
  v8f acc[8];
#pragma unroll
  for (int t = 0; t < 8; ++t) {
    const int n0 = (wave + 8 * t) * 16;
    v8f c = {};
    c = WMMA_BF16(a0, load_bt_bf16(kb, DD, n0, 0, lane), c);
    c = WMMA_BF16(a1, load_bt_bf16(kb, DD, n0, 32, lane), c);
    acc[t] = c;
  }

  // ---- Phase 2: scale + dual mask + softmax ----------------------------
  const float scale = 0.03125f;  // 1/sqrt(DK) = 1/32
  int m1[8];
#pragma unroll
  for (int r = 0; r < 8; ++r) m1[r] = mask1[b * LL + q0 + half * 8 + r];
#pragma unroll
  for (int t = 0; t < 8; ++t) {
    const int m2 = mask2[b * LL + (wave + 8 * t) * 16 + nlane];
#pragma unroll
    for (int r = 0; r < 8; ++r) {
      float v = acc[t][r] * scale;
      if (m1[r] == 0 || m2 == 0) v = NEGV;
      acc[t][r] = v;
    }
  }

  float pm[8];
#pragma unroll
  for (int r = 0; r < 8; ++r) {
    float m = acc[0][r];
#pragma unroll
    for (int t = 1; t < 8; ++t) m = fmaxf(m, acc[t][r]);
    pm[r] = m;
  }
#pragma unroll
  for (int mk = 1; mk < 16; mk <<= 1)
#pragma unroll
    for (int r = 0; r < 8; ++r) pm[r] = fmaxf(pm[r], __shfl_xor(pm[r], mk, 32));
  if (nlane == 0)
#pragma unroll
    for (int r = 0; r < 8; ++r) wmax[wave][half * 8 + r] = pm[r];
  __syncthreads();

  float rm[8];
#pragma unroll
  for (int r = 0; r < 8; ++r) {
    float m = wmax[0][half * 8 + r];
#pragma unroll
    for (int w = 1; w < 8; ++w) m = fmaxf(m, wmax[w][half * 8 + r]);
    rm[r] = m;
  }

  float ps[8] = {0, 0, 0, 0, 0, 0, 0, 0};
#pragma unroll
  for (int t = 0; t < 8; ++t)
#pragma unroll
    for (int r = 0; r < 8; ++r) {
      float p = __expf(acc[t][r] - rm[r]);
      acc[t][r] = p;
      ps[r] += p;
    }
#pragma unroll
  for (int mk = 1; mk < 16; mk <<= 1)
#pragma unroll
    for (int r = 0; r < 8; ++r) ps[r] += __shfl_xor(ps[r], mk, 32);
  if (nlane == 0)
#pragma unroll
    for (int r = 0; r < 8; ++r) wsum[wave][half * 8 + r] = ps[r];
  __syncthreads();

  float rinv[8];
#pragma unroll
  for (int r = 0; r < 8; ++r) {
    float s = 0.f;
#pragma unroll
    for (int w = 0; w < 8; ++w) s += wsum[w][half * 8 + r];
    rinv[r] = 1.0f / s;
  }

  // ---- Phase 3: write fp32 attn (output) + bf16 probs to LDS -----------
#pragma unroll
  for (int t = 0; t < 8; ++t) {
    const int n0 = (wave + 8 * t) * 16;
    v8f d;
#pragma unroll
    for (int r = 0; r < 8; ++r) d[r] = acc[t][r] * rinv[r];
    store_tile_f32(ab, LL, q0, n0, lane, d);
#pragma unroll
    for (int r = 0; r < 8; ++r)
      sprob[(r + 8 * half) * LL + n0 + nlane] = f2bf(d[r]);
  }
  __syncthreads();

  // ---- Phase 4: ctx = probs(16x1024) @ V_head(1024x64) -----------------
  const int nc = wave & 3, kh = wave >> 2;
  v8f cacc = {};
  for (int s = 0; s < 16; ++s) {
    const int kt = kh * 16 + s;
    const int k0 = kt * 32;
    const __bf16* p = sprob + (size_t)nlane * LL + k0 + half * 16;   // A from LDS
    v8bf lo = *(const v8bf*)(p);
    v8bf hi = *(const v8bf*)(p + 8);
    v16bf a = __builtin_shufflevector(lo, hi, 0, 1, 2, 3, 4, 5, 6, 7,
                                      8, 9, 10, 11, 12, 13, 14, 15);
    v16bf bv = *(const v16bf*)(vpk + ((size_t)((b * 32 + kt) * 64 + (h * 4 + nc)) * 512
                                      + lane * 16));
    cacc = WMMA_BF16(a, bv, cacc);
  }
  if (kh == 1) {
#pragma unroll
    for (int r = 0; r < 8; ++r) cred[nc][half * 8 + r][nlane] = cacc[r];
  }
  __syncthreads();
  if (kh == 0) {
#pragma unroll
    for (int r = 0; r < 8; ++r) cacc[r] += cred[nc][half * 8 + r][nlane];
    __bf16* cb = ctxbf + ((size_t)(b * LL + q0 + half * 8)) * DD + h * HD + nc * 16 + nlane;
#pragma unroll
    for (int r = 0; r < 8; ++r) cb[(size_t)r * DD] = f2bf(cacc[r]);
  }
}

// ---------------------------------------------------------------------------
// FBP bilinear gate: one block (64 thr) per row; replicates sign-via-div NaN.
// ---------------------------------------------------------------------------
__global__ __launch_bounds__(64) void fbp_gate(const float* __restrict__ qp,
                                               const float* __restrict__ kp,
                                               const float* __restrict__ Wf1,
                                               const float* __restrict__ Wf2,
                                               const float* __restrict__ Wg,
                                               const float* __restrict__ bg,
                                               float* __restrict__ gate) {
  const int row = blockIdx.x, j = threadIdx.x;  // j in [0,64)
  const float* qr = qp + (size_t)row * DD;
  const float* kr = kp + (size_t)row * DD;
  float y1 = 0.f, y2 = 0.f;
  for (int k = 0; k < DD; ++k) {
    y1 += qr[k] * Wf1[k * 64 + j];
    y2 += kr[k] * Wf2[k * 64 + j];
  }
  __shared__ float s1[64], s2[64], sf[32];
  s1[j] = y1;
  s2[j] = y2;
  __syncthreads();
  if (j < 32)  // avgpool(k=2)*2 == sum of adjacent pair of products
    sf[j] = s1[2 * j] * s2[2 * j] + s1[2 * j + 1] * s2[2 * j + 1];
  __syncthreads();
  if (j == 0) {
    float ss = 0.f;
    for (int hh = 0; hh < 32; ++hh) ss += sf[hh] * sf[hh];
    float nrm = fmaxf(sqrtf(ss), 1e-12f);
    float acc = bg[0];
    for (int hh = 0; hh < 32; ++hh) acc += (sf[hh] / nrm) * Wg[hh];
    float g = tanhf(acc);
    float gs = g / fabsf(g);              // NaN at exactly 0 (matches reference)
    gate[row] = (gs + fabsf(gs)) * 0.5f;  // 1 if >0, 0 if <0, NaN if 0
  }
}

// ---------------------------------------------------------------------------
// out = LayerNorm(fco*gate + residual) * gamma + beta, NaN -> 0
// ---------------------------------------------------------------------------
__global__ __launch_bounds__(256) void gated_ln(const float* __restrict__ fco,
                                                const float* __restrict__ gate,
                                                const float* __restrict__ resid,
                                                const float* __restrict__ gamma,
                                                const float* __restrict__ beta,
                                                float* __restrict__ out) {
  const int row = blockIdx.x, tid = threadIdx.x;
  const float g = gate[row];
  float v[4];
#pragma unroll
  for (int i = 0; i < 4; ++i) {
    const int c = tid + 256 * i;
    v[i] = fco[(size_t)row * DD + c] * g + resid[(size_t)row * DD + c];
  }
  __shared__ float red[256];
  red[tid] = v[0] + v[1] + v[2] + v[3];
  __syncthreads();
  for (int o = 128; o > 0; o >>= 1) {
    if (tid < o) red[tid] += red[tid + o];
    __syncthreads();
  }
  const float mu = red[0] * (1.0f / DD);
  __syncthreads();
  float s2 = 0.f;
#pragma unroll
  for (int i = 0; i < 4; ++i) {
    const float d = v[i] - mu;
    s2 += d * d;
  }
  red[tid] = s2;
  __syncthreads();
  for (int o = 128; o > 0; o >>= 1) {
    if (tid < o) red[tid] += red[tid + o];
    __syncthreads();
  }
  const float rstd = rsqrtf(red[0] * (1.0f / DD) + 1e-6f);
#pragma unroll
  for (int i = 0; i < 4; ++i) {
    const int c = tid + 256 * i;
    float o_ = (v[i] - mu) * rstd * gamma[c] + beta[c];
    if (o_ != o_) o_ = 0.f;  // NaN masking
    out[(size_t)row * DD + c] = o_;
  }
}

// ---------------------------------------------------------------------------
// Launch. Workspace layout (byte offsets, total 128MB + 16KB):
//   0MB   qp   fp32 (16MB)      64MB  qbf  bf16 (8MB)    104MB vpk   bf16 (8MB)
//   16MB  kp   fp32 (16MB)      72MB  kbf  bf16 (8MB)    112MB ctxbf bf16 (8MB)
//   32MB  vp   fp32 (16MB)      80MB  vbf  bf16 (8MB)    120MB Wq/Wk/Wv/Wfc packed (4x2MB)
//   48MB  fco  fp32 (16MB)      88MB  qpbf bf16 (8MB)    128MB gate  fp32 (16KB)
//                               96MB  kpbf bf16 (8MB)
// d_out = [ out (B*L*D) | attn (B*H*L*L) ]
// ---------------------------------------------------------------------------
extern "C" void kernel_launch(void* const* d_in, const int* in_sizes, int n_in,
                              void* d_out, int out_size, void* d_ws, size_t ws_size,
                              hipStream_t stream) {
  const float* q     = (const float*)d_in[0];
  const float* k     = (const float*)d_in[1];
  const float* v     = (const float*)d_in[2];
  const int*   mask1 = (const int*)d_in[3];
  const int*   mask2 = (const int*)d_in[4];
  const float* Wq    = (const float*)d_in[5];
  const float* Wk    = (const float*)d_in[6];
  const float* Wv    = (const float*)d_in[7];
  const float* Wfc   = (const float*)d_in[8];
  const float* Wf1   = (const float*)d_in[9];
  const float* Wf2   = (const float*)d_in[10];
  const float* Wg    = (const float*)d_in[11];
  const float* bg    = (const float*)d_in[12];
  const float* gamma = (const float*)d_in[13];
  const float* beta  = (const float*)d_in[14];

  float* out  = (float*)d_out;
  float* attn = out + (size_t)BB * LL * DD;

  const size_t MB = 1024 * 1024;
  char* w = (char*)d_ws;
  float*  qp    = (float*)(w + 0 * MB);
  float*  kp    = (float*)(w + 16 * MB);
  float*  vp    = (float*)(w + 32 * MB);
  float*  fco   = (float*)(w + 48 * MB);
  __bf16* qbf   = (__bf16*)(w + 64 * MB);
  __bf16* kbf   = (__bf16*)(w + 72 * MB);
  __bf16* vbf   = (__bf16*)(w + 80 * MB);
  __bf16* qpbf  = (__bf16*)(w + 88 * MB);
  __bf16* kpbf  = (__bf16*)(w + 96 * MB);
  __bf16* vpk   = (__bf16*)(w + 104 * MB);
  __bf16* ctxbf = (__bf16*)(w + 112 * MB);
  __bf16* Wqp   = (__bf16*)(w + 120 * MB);
  __bf16* Wkp   = (__bf16*)(w + 122 * MB);
  __bf16* Wvp   = (__bf16*)(w + 124 * MB);
  __bf16* Wfcp  = (__bf16*)(w + 126 * MB);
  float*  gate  = (float*)(w + 128 * MB);

  const int CVTG = (BB * LL * DD) / 4 / 256;   // 4096 blocks
  cvt_f32_bf16<<<CVTG, 256, 0, stream>>>(q, qbf);
  cvt_f32_bf16<<<CVTG, 256, 0, stream>>>(k, kbf);
  cvt_f32_bf16<<<CVTG, 256, 0, stream>>>(v, vbf);

  pack_b<<<256, 256, 0, stream>>>(Wq, Wqp, DD);    // (1024/32)*(1024/16)/8
  pack_b<<<256, 256, 0, stream>>>(Wk, Wkp, DD);
  pack_b<<<256, 256, 0, stream>>>(Wv, Wvp, DD);
  pack_b<<<256, 256, 0, stream>>>(Wfc, Wfcp, DD);

  const dim3 gg(DD / 128, (BB * LL) / 64);   // 2x2 reg-tiled: block = 64x128
  gemm_bf16p<<<gg, 256, 0, stream>>>(qbf, Wqp, qp, qpbf);
  gemm_bf16p<<<gg, 256, 0, stream>>>(kbf, Wkp, kp, kpbf);
  gemm_bf16p<<<gg, 256, 0, stream>>>(vbf, Wvp, vp, (__bf16*)nullptr);

  pack_b<<<1024, 256, 0, stream>>>(vp, vpk, DD);   // vp as (B*L) x D, tiles per batch

  fbp_gate<<<BB * LL, 64, 0, stream>>>(qp, kp, Wf1, Wf2, Wg, bg, gate);

  attn_fused<<<dim3(LL / 16, BB * HH), 256, 0, stream>>>(qpbf, kpbf, mask1, mask2,
                                                         attn, vpk, ctxbf);

  gemm_bf16p<<<gg, 256, 0, stream>>>(ctxbf, Wfcp, fco, (__bf16*)nullptr);
  gated_ln<<<BB * LL, 256, 0, stream>>>(fco, gate, q, gamma, beta, out);
}
